// SGCN_747324309855
// MI455X (gfx1250) — compile-verified, matrix operand order checked by reading, not run on hardware
//
#include <hip/hip_runtime.h>

#define DD 64

typedef __attribute__((ext_vector_type(16))) __bf16 v16bf;
typedef __attribute__((ext_vector_type(8)))  __bf16 v8bf;
typedef __attribute__((ext_vector_type(8)))  float  v8f;

// ---------------- kernel: zero u32 array ----------------
__global__ void k_zero_u32(unsigned* __restrict__ p, int n) {
  int i = blockIdx.x * blockDim.x + threadIdx.x;
  if (i < n) p[i] = 0u;
}

// ---------------- kernel: fused q/k/v/skip projections via bf16 WMMA ----------------
// One wave (32 lanes) computes a 16-row tile of all four 16x64 projections.
// A (16x32 bf16) built per ISA layout; B fragments come from an LDS-resident
// transposed bf16 copy of the weights so each half-fragment is one 16B ds load.
__global__ __launch_bounds__(256) void k_proj(
    const float* __restrict__ x,
    const float* __restrict__ Wq, const float* __restrict__ bq,
    const float* __restrict__ Wk, const float* __restrict__ bk,
    const float* __restrict__ Wv, const float* __restrict__ bv,
    const float* __restrict__ Wsk, const float* __restrict__ bsk,
    float* __restrict__ q, float* __restrict__ k, float* __restrict__ v,
    float* __restrict__ skip, int N, int nTiles)
{
  __shared__ alignas(16) __bf16 Wt[4][DD][DD];   // Wt[w][n][kk] = W_w[kk][n]  (32 KB)
  {
    const float* Wsrc[4] = {Wq, Wk, Wv, Wsk};
#pragma unroll
    for (int w = 0; w < 4; ++w)
      for (int idx = threadIdx.x; idx < DD * DD; idx += 256) {
        int kk = idx >> 6, n = idx & 63;
        Wt[w][n][kk] = (__bf16)Wsrc[w][idx];
      }
  }
  __syncthreads();

  const int wave = threadIdx.x >> 5;
  const int lane = threadIdx.x & 31;
  const int tile = blockIdx.x * 8 + wave;
  if (tile >= nTiles) return;
  const int row0 = tile * 16;
  const int h = lane >> 4;       // half-wave selector
  const int m = lane & 15;       // A row / B-D column within 16
  const bool fullTile = (row0 + 16 <= N);   // wave-uniform

  // ---- A fragments: rows of x converted to bf16, ISA 16-bit A layout ----
  int rA = row0 + m; if (rA > N - 1) rA = N - 1;
  const float* xrow = x + (size_t)rA * DD;
  v16bf a[2];
#pragma unroll
  for (int t = 0; t < 2; ++t) {
    const int k0 = 32 * t + 8 * h;        // elements 0..7
    const int k1 = 32 * t + 16 + 8 * h;   // elements 8..15
#pragma unroll
    for (int e = 0; e < 8; ++e) {
      a[t][e]     = (__bf16)xrow[k0 + e];
      a[t][8 + e] = (__bf16)xrow[k1 + e];
    }
  }

  const float* biases[4] = {bq, bk, bv, bsk};
  float* outs[4] = {q, k, v, skip};
#pragma unroll
  for (int w = 0; w < 4; ++w) {
#pragma unroll
    for (int c = 0; c < 4; ++c) {
      const int n = c * 16 + m;
      const v8bf* wrow = (const v8bf*)(&Wt[w][n][0]);  // 8 chunks of 8 bf16
      v16bf b[2];
#pragma unroll
      for (int t = 0; t < 2; ++t) {
        v8bf lo = wrow[(32 * t + 8 * h) >> 3];
        v8bf hi = wrow[(32 * t + 16 + 8 * h) >> 3];
#pragma unroll
        for (int e = 0; e < 8; ++e) { b[t][e] = lo[e]; b[t][8 + e] = hi[e]; }
      }
      const float bias = biases[w][n];
      v8f acc;
#pragma unroll
      for (int r = 0; r < 8; ++r) acc[r] = bias;
      acc = __builtin_amdgcn_wmma_f32_16x16x32_bf16(false, a[0], false, b[0],
                                                    (short)0, acc, false, false);
      acc = __builtin_amdgcn_wmma_f32_16x16x32_bf16(false, a[1], false, b[1],
                                                    (short)0, acc, false, false);
      // per-lane base pointer; rows become immediate-offset stores (r*256 B)
      float* op = outs[w] + ((size_t)(row0 + 8 * h) * DD + n);
      if (fullTile) {
#pragma unroll
        for (int r = 0; r < 8; ++r) op[r * DD] = acc[r];
      } else {
#pragma unroll
        for (int r = 0; r < 8; ++r)
          if (row0 + 8 * h + r < N) op[r * DD] = acc[r];
      }
    }
  }
}

// ---------------- kernel: in-degree histogram ----------------
__global__ void k_degree(const int* __restrict__ ei, unsigned* __restrict__ deg,
                         int E) {
  int e = blockIdx.x * blockDim.x + threadIdx.x;
  if (e < E) atomicAdd(&deg[ei[E + e]], 1u);
}

// ---------------- kernel: single-block exclusive scan (offsets + cursor) ----------------
__global__ __launch_bounds__(1024) void k_scan(const unsigned* __restrict__ deg,
                                               unsigned* __restrict__ offsets,
                                               unsigned* __restrict__ cursor,
                                               int n) {
  __shared__ unsigned tmp[1024];
  __shared__ unsigned carry;
  if (threadIdx.x == 0) carry = 0;
  __syncthreads();
  for (int base = 0; base < n; base += 1024) {
    const int i = base + (int)threadIdx.x;
    const unsigned val = (i < n) ? deg[i] : 0u;
    tmp[threadIdx.x] = val;
    __syncthreads();
#pragma unroll
    for (int off = 1; off < 1024; off <<= 1) {
      unsigned t = (threadIdx.x >= (unsigned)off) ? tmp[threadIdx.x - off] : 0u;
      __syncthreads();
      tmp[threadIdx.x] += t;
      __syncthreads();
    }
    const unsigned excl = tmp[threadIdx.x] - val;
    if (i < n) {
      const unsigned o = carry + excl;
      offsets[i] = o;
      cursor[i] = o;
    }
    __syncthreads();
    if (threadIdx.x == 1023) carry += tmp[1023];
    __syncthreads();
  }
  if (threadIdx.x == 0) offsets[n] = carry;
}

// ---------------- kernel: scatter edges into CSR buckets by dst ----------------
__global__ void k_scatter(const int* __restrict__ ei, unsigned* __restrict__ cursor,
                          int* __restrict__ csr_src, int E) {
  int e = blockIdx.x * blockDim.x + threadIdx.x;
  if (e < E) {
    const int s = ei[e];
    const int d = ei[E + e];
    const unsigned pos = atomicAdd(&cursor[d], 1u);
    csr_src[pos] = s;
  }
}

// ---------------- kernel: fused online-softmax attention aggregation ----------------
// One wave per destination node; lane j owns features j and j+32.
// Atomic-free: segment max/sum/weighted-sum are node-local (flash-style).
__global__ __launch_bounds__(256) void k_attn(
    const float* __restrict__ q, const float* __restrict__ k,
    const float* __restrict__ v, const unsigned* __restrict__ offsets,
    const int* __restrict__ csr_src, float* __restrict__ out, int N) {
  const int wave = threadIdx.x >> 5;
  const int lane = threadIdx.x & 31;
  const int node = blockIdx.x * 8 + wave;
  if (node >= N) return;
  const unsigned beg = offsets[node], end = offsets[node + 1];
  if (beg == end) return;  // out already holds the skip projection

  const float scale = 0.125f;  // 1/sqrt(64)
  const float q0 = q[(size_t)node * DD + lane];
  const float q1 = q[(size_t)node * DD + 32 + lane];

  float mx = -3.4e38f, s = 0.0f, acc0 = 0.0f, acc1 = 0.0f;
  int src_next = csr_src[beg];
  for (unsigned e = beg; e < end; ++e) {
    const int src = src_next;
    if (e + 1 < end) {
      src_next = csr_src[e + 1];
      __builtin_prefetch((const void*)(k + (size_t)src_next * DD), 0, 3);
      __builtin_prefetch((const void*)(v + (size_t)src_next * DD), 0, 3);
    }
    const float* krow = k + (size_t)src * DD;
    const float* vrow = v + (size_t)src * DD;
    float d = q0 * krow[lane] + q1 * krow[32 + lane];
#pragma unroll
    for (int off = 16; off > 0; off >>= 1) d += __shfl_xor(d, off, 32);
    d *= scale;
    const float mnew = fmaxf(mx, d);
    const float corr = __expf(mx - mnew);
    const float ex = __expf(d - mnew);
    s = s * corr + ex;
    acc0 = acc0 * corr + ex * vrow[lane];
    acc1 = acc1 * corr + ex * vrow[32 + lane];
    mx = mnew;
  }
  const float inv = 1.0f / (s + 1e-16f);
  out[(size_t)node * DD + lane] += acc0 * inv;
  out[(size_t)node * DD + 32 + lane] += acc1 * inv;
}

extern "C" void kernel_launch(void* const* d_in, const int* in_sizes, int n_in,
                              void* d_out, int out_size, void* d_ws, size_t ws_size,
                              hipStream_t stream) {
  const float* x  = (const float*)d_in[0];
  const int*   ei = (const int*)d_in[1];   // [2][E]: row0=src, row1=dst
  // d_in[2] = edge_type (unused by the reference forward)
  const float* Wq  = (const float*)d_in[3];
  const float* bq  = (const float*)d_in[4];
  const float* Wk  = (const float*)d_in[5];
  const float* bk  = (const float*)d_in[6];
  const float* Wv  = (const float*)d_in[7];
  const float* bv  = (const float*)d_in[8];
  const float* Wsk = (const float*)d_in[9];
  const float* bsk = (const float*)d_in[10];
  float* out = (float*)d_out;

  const int N = in_sizes[0] / DD;
  const int E = in_sizes[2];

  // workspace carve-out (256B aligned slices)
  size_t off = 0;
  char* base = (char*)d_ws;
  auto take = [&](size_t bytes) -> void* {
    void* p = base + off;
    off += (bytes + 255) & ~(size_t)255;
    return p;
  };
  float*    q       = (float*)take((size_t)N * DD * sizeof(float));
  float*    k       = (float*)take((size_t)N * DD * sizeof(float));
  float*    v       = (float*)take((size_t)N * DD * sizeof(float));
  unsigned* deg     = (unsigned*)take((size_t)N * sizeof(unsigned));
  unsigned* offsets = (unsigned*)take((size_t)(N + 1) * sizeof(unsigned));
  unsigned* cursor  = (unsigned*)take((size_t)N * sizeof(unsigned));
  int*      csr_src = (int*)take((size_t)E * sizeof(int));
  (void)ws_size; (void)n_in;

  const int nTiles = (N + 15) / 16;

  k_zero_u32<<<(N + 255) / 256, 256, 0, stream>>>(deg, N);
  k_proj<<<(nTiles + 7) / 8, 256, 0, stream>>>(x, Wq, bq, Wk, bk, Wv, bv, Wsk, bsk,
                                               q, k, v, out, N, nTiles);
  k_degree<<<(E + 255) / 256, 256, 0, stream>>>(ei, deg, E);
  k_scan<<<1, 1024, 0, stream>>>(deg, offsets, cursor, N);
  k_scatter<<<(E + 255) / 256, 256, 0, stream>>>(ei, cursor, csr_src, E);
  k_attn<<<(N + 7) / 8, 256, 0, stream>>>(q, k, v, offsets, csr_src, out, N);
}
